// QKVAttention_67293547593871
// MI455X (gfx1250) — matrix operand order repaired and checked by reading, not compiled, not run on hardware
//
#include <hip/hip_runtime.h>

typedef __attribute__((ext_vector_type(16))) _Float16 v16h;
typedef __attribute__((ext_vector_type(8)))  _Float16 v8h;
typedef __attribute__((ext_vector_type(4)))  _Float16 v4h;
typedef __attribute__((ext_vector_type(2)))  _Float16 v2h;
typedef __attribute__((ext_vector_type(8)))  float    v8f;
typedef __attribute__((ext_vector_type(4)))  float    v4f;

#define N_HEADS  16
#define T_LEN    2048
#define CH       64
#define ROW_W    (3 * N_HEADS * CH)   // 3072 floats per qkv row
#define BM       128                  // query rows per workgroup (4 waves x 2 strips x 16)
#define BN       64                   // key rows per LDS tile
#define NWAVE    4
#define NTHREADS 128

// packed f32->f16 convert (v_cvt_pk_rtz_f16_f32), bit_cast from __fp16x2 to _Float16x2
__device__ __forceinline__ v2h pk_h2(float a, float b) {
    return __builtin_bit_cast(v2h, __builtin_amdgcn_cvt_pkrtz(a, b));
}

// ---- DPP 16-lane row reductions (pure VALU, no LDS/bpermute) ----
template <int CTRL>
__device__ __forceinline__ float dpp_mov(float x) {
    return __builtin_bit_cast(float,
        __builtin_amdgcn_update_dpp(0, __builtin_bit_cast(int, x),
                                    CTRL, 0xF, 0xF, true));
}
__device__ __forceinline__ float rowmax16(float x) {
    x = fmaxf(x, dpp_mov<0xB1>(x));   // quad_perm(1,0,3,2)  : xor 1
    x = fmaxf(x, dpp_mov<0x4E>(x));   // quad_perm(2,3,0,1)  : xor 2
    x = fmaxf(x, dpp_mov<0x141>(x));  // row_half_mirror
    x = fmaxf(x, dpp_mov<0x140>(x));  // row_mirror
    return x;
}
__device__ __forceinline__ float rowsum16(float x) {
    x += dpp_mov<0xB1>(x);
    x += dpp_mov<0x4E>(x);
    x += dpp_mov<0x141>(x);
    x += dpp_mov<0x140>(x);
    return x;
}

// Flash-attention: head-slice per blockIdx.x, 128-row query tile per blockIdx.y.
// f16 WMMA, f32 accumulate, online softmax in log2 domain.
__global__ __launch_bounds__(NTHREADS)
void qkv_attn_fa_kernel(const float* __restrict__ qkv, float* __restrict__ out)
{
    __shared__ __align__(32) _Float16 sK [BN][CH];          // K tile (key, ch)
    __shared__ __align__(32) _Float16 sVt[CH][BN];          // V transposed (ch, key)
    __shared__ __align__(32) _Float16 sP [NWAVE][32][BN];   // per-wave P staging, 2 strips

    const int g    = blockIdx.x;
    const int qt   = blockIdx.y;
    const int b    = g >> 4;
    const int hh   = g & 15;
    const int tid  = threadIdx.x;
    const int wv   = tid >> 5;
    const int lane = tid & 31;
    const int lm   = lane & 15;
    const int half = lane >> 4;

    const size_t bbase = (size_t)b * ((size_t)T_LEN * ROW_W);
    const int    i0    = qt * BM;

    // ---- Q A-fragments for two 16-row strips; scale folds softmax 1/8 and log2(e) ----
    v16h qa[2][2];
    {
        const float qs = 0.125f * 1.44269504088896340736f;
        #pragma unroll
        for (int s = 0; s < 2; ++s) {
            const int qrow = i0 + wv * 32 + s * 16 + lm;
            const size_t ra = bbase + (size_t)(hh * 128 + (qrow >> 4)) * ROW_W
                                    + (size_t)((qrow & 15) * CH);
            #pragma unroll
            for (int f = 0; f < 2; ++f) {
                const int c0 = f * 32 + half * 8;
                const int c1 = f * 32 + 16 + half * 8;
                const v4f x0 = *(const v4f*)(qkv + ra + c0);
                const v4f x1 = *(const v4f*)(qkv + ra + c0 + 4);
                const v4f x2 = *(const v4f*)(qkv + ra + c1);
                const v4f x3 = *(const v4f*)(qkv + ra + c1 + 4);
                v2h p;
                p = pk_h2(x0[0]*qs, x0[1]*qs); qa[s][f][0]=p[0];  qa[s][f][1]=p[1];
                p = pk_h2(x0[2]*qs, x0[3]*qs); qa[s][f][2]=p[0];  qa[s][f][3]=p[1];
                p = pk_h2(x1[0]*qs, x1[1]*qs); qa[s][f][4]=p[0];  qa[s][f][5]=p[1];
                p = pk_h2(x1[2]*qs, x1[3]*qs); qa[s][f][6]=p[0];  qa[s][f][7]=p[1];
                p = pk_h2(x2[0]*qs, x2[1]*qs); qa[s][f][8]=p[0];  qa[s][f][9]=p[1];
                p = pk_h2(x2[2]*qs, x2[3]*qs); qa[s][f][10]=p[0]; qa[s][f][11]=p[1];
                p = pk_h2(x3[0]*qs, x3[1]*qs); qa[s][f][12]=p[0]; qa[s][f][13]=p[1];
                p = pk_h2(x3[2]*qs, x3[3]*qs); qa[s][f][14]=p[0]; qa[s][f][15]=p[1];
            }
        }
    }

    // ---- online-softmax state (log2 domain), C-frag row = half*8 + r ----
    float m_row[2][8], l_row[2][8];
    v8f   oacc[2][4];
    #pragma unroll
    for (int s = 0; s < 2; ++s) {
        #pragma unroll
        for (int r = 0; r < 8; ++r) { m_row[s][r] = -1e30f; l_row[s][r] = 0.0f; }
        #pragma unroll
        for (int c = 0; c < 4; ++c) { v8f z = {}; oacc[s][c] = z; }
    }

    for (int kt = 0; kt < T_LEN / BN; ++kt) {
        const int j0 = kt * BN;
        __syncthreads();

        // ---- cooperative K/V tile staging: 2 threads per key row, packed converts ----
        {
            const int n    = tid >> 1;
            const int hsel = tid & 1;
            const int j    = j0 + n;
            const size_t ra = bbase + (size_t)(hh * 128 + (j >> 4)) * ROW_W
                                    + (size_t)((j & 15) * CH);
            const float* kptr = qkv + ra + 1024 + hsel * 32;
            const float* vptr = qkv + ra + 2048 + hsel * 32;
            #pragma unroll
            for (int q4 = 0; q4 < 8; ++q4) {
                const int cb = hsel * 32 + q4 * 4;
                const v4f kx = ((const v4f*)kptr)[q4];
                const v2h k01 = pk_h2(kx[0], kx[1]);
                const v2h k23 = pk_h2(kx[2], kx[3]);
                v4h kh; kh[0]=k01[0]; kh[1]=k01[1]; kh[2]=k23[0]; kh[3]=k23[1];
                *(v4h*)&sK[n][cb] = kh;
                const v4f vx = ((const v4f*)vptr)[q4];
                const v2h v01 = pk_h2(vx[0], vx[1]);
                const v2h v23 = pk_h2(vx[2], vx[3]);
                sVt[cb + 0][n] = v01[0];
                sVt[cb + 1][n] = v01[1];
                sVt[cb + 2][n] = v23[0];
                sVt[cb + 3][n] = v23[1];
            }
        }
        __syncthreads();

        // ---- S = Q K^T : B-fragments shared across both query strips ----
        v8f st[2][4];
        #pragma unroll
        for (int s = 0; s < 2; ++s)
            #pragma unroll
            for (int jt = 0; jt < 4; ++jt) { v8f z = {}; st[s][jt] = z; }
        #pragma unroll
        for (int f = 0; f < 2; ++f) {
            #pragma unroll
            for (int jt = 0; jt < 4; ++jt) {
                const v8h k0 = *(const v8h*)&sK[jt * 16 + lm][f * 32 + half * 8];
                const v8h k1 = *(const v8h*)&sK[jt * 16 + lm][f * 32 + 16 + half * 8];
                v16h bk;
                #pragma unroll
                for (int e = 0; e < 8; ++e) { bk[e] = k0[e]; bk[8 + e] = k1[e]; }
                st[0][jt] = __builtin_amdgcn_wmma_f32_16x16x32_f16(
                                false, qa[0][f], false, bk, (short)0, st[0][jt], false, false);
                st[1][jt] = __builtin_amdgcn_wmma_f32_16x16x32_f16(
                                false, qa[1][f], false, bk, (short)0, st[1][jt], false, false);
            }
        }

        // ---- online softmax (log2 domain, DPP reductions) + P staging ----
        #pragma unroll
        for (int s = 0; s < 2; ++s) {
            #pragma unroll
            for (int r = 0; r < 8; ++r) {
                float mx = fmaxf(fmaxf(st[s][0][r], st[s][1][r]),
                                 fmaxf(st[s][2][r], st[s][3][r]));
                mx = rowmax16(mx);
                const float mnew = fmaxf(m_row[s][r], mx);
                const float a    = __builtin_amdgcn_exp2f(m_row[s][r] - mnew);
                m_row[s][r] = mnew;
                float sum = 0.0f;
                #pragma unroll
                for (int jt = 0; jt < 4; ++jt) {
                    const float p = __builtin_amdgcn_exp2f(st[s][jt][r] - mnew);
                    st[s][jt][r] = p;
                    sum += p;
                }
                sum = rowsum16(sum);
                l_row[s][r] = l_row[s][r] * a + sum;
                #pragma unroll
                for (int c = 0; c < 4; ++c)
                    oacc[s][c][r] *= a;
            }
            #pragma unroll
            for (int jt = 0; jt < 4; ++jt) {
                #pragma unroll
                for (int r = 0; r < 8; r += 2) {
                    const v2h p2 = pk_h2(st[s][jt][r], st[s][jt][r + 1]);
                    sP[wv][s * 16 + half * 8 + r    ][jt * 16 + lm] = p2[0];
                    sP[wv][s * 16 + half * 8 + r + 1][jt * 16 + lm] = p2[1];
                }
            }
        }
        __syncthreads();

        // ---- O += P V : V B-fragments shared across both strips ----
        #pragma unroll
        for (int f = 0; f < 2; ++f) {
            v16h pa[2];
            #pragma unroll
            for (int s = 0; s < 2; ++s) {
                const v8h p0 = *(const v8h*)&sP[wv][s * 16 + lm][f * 32 + half * 8];
                const v8h p1 = *(const v8h*)&sP[wv][s * 16 + lm][f * 32 + 16 + half * 8];
                #pragma unroll
                for (int e = 0; e < 8; ++e) { pa[s][e] = p0[e]; pa[s][8 + e] = p1[e]; }
            }
            #pragma unroll
            for (int c = 0; c < 4; ++c) {
                const v8h v0 = *(const v8h*)&sVt[c * 16 + lm][f * 32 + half * 8];
                const v8h v1 = *(const v8h*)&sVt[c * 16 + lm][f * 32 + 16 + half * 8];
                v16h bv;
                #pragma unroll
                for (int e = 0; e < 8; ++e) { bv[e] = v0[e]; bv[8 + e] = v1[e]; }
                oacc[0][c] = __builtin_amdgcn_wmma_f32_16x16x32_f16(
                                 false, pa[0], false, bv, (short)0, oacc[0][c], false, false);
                oacc[1][c] = __builtin_amdgcn_wmma_f32_16x16x32_f16(
                                 false, pa[1], false, bv, (short)0, oacc[1][c], false, false);
            }
        }
    }

    // ---- epilogue: O / l, contiguous [64, 2048, 64] output ----
    #pragma unroll
    for (int s = 0; s < 2; ++s) {
        #pragma unroll
        for (int c = 0; c < 4; ++c) {
            #pragma unroll
            for (int r = 0; r < 8; ++r) {
                const int i   = i0 + wv * 32 + s * 16 + half * 8 + r;
                const int col = c * 16 + lm;
                out[(size_t)g * (T_LEN * CH) + (size_t)i * CH + col] =
                    oacc[s][c][r] / l_row[s][r];
            }
        }
    }
}

extern "C" void kernel_launch(void* const* d_in, const int* in_sizes, int n_in,
                              void* d_out, int out_size, void* d_ws, size_t ws_size,
                              hipStream_t stream) {
    (void)n_in; (void)out_size; (void)d_ws; (void)ws_size;
    const float* qkv = (const float*)d_in[0];
    float* out = (float*)d_out;
    const int bs = in_sizes[0] / (T_LEN * ROW_W);   // = 4
    dim3 grid(bs * N_HEADS, T_LEN / BM);            // 64 head-slices x 16 query tiles
    qkv_attn_fa_kernel<<<grid, NTHREADS, 0, stream>>>(qkv, out);
}
